// Attention_3556232921308
// MI455X (gfx1250) — compile-verified, matrix operand order checked by reading, not run on hardware
//
#include <hip/hip_runtime.h>

// ---------------------------------------------------------------------------
// Types
// ---------------------------------------------------------------------------
typedef __bf16 bf16_t;
typedef __attribute__((ext_vector_type(16))) __bf16 v16bf;
typedef __attribute__((ext_vector_type(8)))  __bf16 v8bf;
typedef __attribute__((ext_vector_type(8)))  float  v8f;
typedef __attribute__((ext_vector_type(4)))  unsigned int v4u;
typedef __attribute__((ext_vector_type(8)))  int v8i;
typedef __attribute__((ext_vector_type(4)))  int v4i;

#define B_SZ   2
#define S_LEN  2048
#define HID    2048
#define NH     16
#define NKV    4
#define HD     128
#define QG_N   4096   // 2*NH*HD
#define KV_N   512    // NKV*HD
#define BS     (B_SZ * S_LEN)   // 4096

// ---------------------------------------------------------------------------
// WMMA helper: D = A(16x32 bf16) * B(32x16 bf16) + C(16x16 f32)
// ---------------------------------------------------------------------------
__device__ __forceinline__ v8f wmma_bf16(v16bf a, v16bf b, v8f c) {
  return __builtin_amdgcn_wmma_f32_16x16x32_bf16(
      /*neg_a=*/false, a, /*neg_b=*/false, b,
      /*c_mod=*/(short)0, c, /*reuse_a=*/false, /*reuse_b=*/false);
}

// Fragment load (row-major source, rows contiguous along K).
// Lane L holds row (L&15); element e holds K = k0 + (e>=8?16:0) + 8*(L>>4) + (e&7).
// Two 16-byte loads per lane.
__device__ __forceinline__ v16bf ld_frag(const bf16_t* p, int ld, int k0) {
  const int lane = threadIdx.x & 31;
  const int row  = lane & 15;
  const int kb   = (lane >> 4) << 3;
  const bf16_t* r = p + (size_t)row * (size_t)ld + k0 + kb;
  union { v16bf v; v8bf h[2]; } u;
  u.h[0] = *(const v8bf*)(r);
  u.h[1] = *(const v8bf*)(r + 16);
  return u.v;
}

// ---------------------------------------------------------------------------
// TDM: stage a 2D bf16 tile (d1 rows of d0 contiguous elems, row stride
// `stride` elems) into LDS. Descriptor per CDNA5 ISA.
// Wave-uniform arguments only. 6-arg builtin (clang-23 / therock-10.0).
// ---------------------------------------------------------------------------
__device__ __forceinline__ void tdm_load_tile(bf16_t* lds_dst, const bf16_t* gsrc,
                                              unsigned d0, unsigned d1, unsigned stride) {
#if __has_builtin(__builtin_amdgcn_tensor_load_to_lds)
  const unsigned long long ga = (unsigned long long)gsrc;
  v4u g0;
  g0[0] = 1u;                                    // count=1, user mode
  g0[1] = (unsigned)(unsigned long long)lds_dst; // LDS byte offset (low 32 bits)
  g0[2] = (unsigned)ga;                          // global_addr[31:0]
  g0[3] = ((unsigned)(ga >> 32) & 0x01FFFFFFu) | (2u << 30); // addr[56:32] | type=2
  v8i g1;
  g1[0] = 0x00010000;                            // workgroup_mask=0, data_size=1 (2B)
  g1[1] = (int)((d0 & 0xFFFFu) << 16);           // tensor_dim0 lo16 (bits 63:48)
  g1[2] = (int)(((d0 >> 16) & 0xFFFFu) | ((d1 & 0xFFFFu) << 16)); // dim0 hi | dim1 lo
  g1[3] = (int)(((d1 >> 16) & 0xFFFFu) | (d0 << 16));             // dim1 hi | tile_dim0
  g1[4] = (int)(d1 & 0xFFFFu);                   // tile_dim1; tile_dim2 = 0
  g1[5] = (int)stride;                           // tensor_dim0_stride (bits 191:160)
  g1[6] = 0;                                     // stride0 hi, stride1 lo
  g1[7] = 0;                                     // stride1 hi
  v4i g2 = {0, 0, 0, 0};                         // dims 2/3 unused (2D)
  v4i g3 = {0, 0, 0, 0};                         // dims 3/4 unused
  v8i g4 = {0, 0, 0, 0, 0, 0, 0, 0};
  __builtin_amdgcn_tensor_load_to_lds(g0, g1, g2, g3, g4, 0);
#else
  // Fallback: wave-cooperative 16B-chunk copy.
  const int lane = threadIdx.x & 31;
  const int cpr  = d0 >> 3;                      // 8-elem chunks per row
  for (int c = lane; c < (int)(d0 * d1) / 8; c += 32) {
    const int row = c / cpr;
    const int col = (c - row * cpr) << 3;
    *(v8bf*)(lds_dst + row * d0 + col) = *(const v8bf*)(gsrc + (size_t)row * stride + col);
  }
#endif
}

__device__ __forceinline__ void tdm_wait_le2() {
#if __has_builtin(__builtin_amdgcn_s_wait_tensorcnt)
  __builtin_amdgcn_s_wait_tensorcnt(2);   // allow next tile's 2 loads in flight
#endif
}
__device__ __forceinline__ void tdm_wait0() {
#if __has_builtin(__builtin_amdgcn_s_wait_tensorcnt)
  __builtin_amdgcn_s_wait_tensorcnt(0);
#endif
}

// ---------------------------------------------------------------------------
// 1. RMSNorm: fp32 hidden -> bf16 x   (weight parameterized as 1 + w)
// ---------------------------------------------------------------------------
__global__ __launch_bounds__(256) void rmsnorm_bf16_kernel(
    const float* __restrict__ hs, const float* __restrict__ w,
    bf16_t* __restrict__ xb) {
  const int row = blockIdx.x;                 // B*S rows
  const float* p = hs + (size_t)row * HID;
  float ss = 0.f;
  for (int j = threadIdx.x; j < HID; j += 256) { float t = p[j]; ss += t * t; }
#pragma unroll
  for (int off = 16; off > 0; off >>= 1) ss += __shfl_xor(ss, off);
  __shared__ float wsum[8];
  if ((threadIdx.x & 31) == 0) wsum[threadIdx.x >> 5] = ss;
  __syncthreads();
  float tot = 0.f;
#pragma unroll
  for (int i = 0; i < 8; ++i) tot += wsum[i];
  const float inv = rsqrtf(tot / (float)HID + 1e-6f);
  bf16_t* o = xb + (size_t)row * HID;
  for (int j = threadIdx.x; j < HID; j += 256)
    o[j] = (bf16_t)(p[j] * inv * (1.0f + w[j]));
}

// ---------------------------------------------------------------------------
// 2. fp32 -> bf16 conversion (weights)
// ---------------------------------------------------------------------------
__global__ __launch_bounds__(256) void f32_to_bf16_kernel(
    const float* __restrict__ in, bf16_t* __restrict__ out, long long n) {
  long long i = (long long)blockIdx.x * 256 + threadIdx.x;
  const long long stride = (long long)gridDim.x * 256;
  for (; i < n; i += stride) out[i] = (bf16_t)in[i];
}

// ---------------------------------------------------------------------------
// 3. GEMM: C[M,N] = A[M,K](bf16) @ W[N,K]^T(bf16); one 64x64 tile per wave.
//    TRANSP=1 stores C^T (N x M) in bf16 — used for V so attention reads
//    contiguous key runs.
// ---------------------------------------------------------------------------
template <int STORE_BF16, int TRANSP>
__global__ __launch_bounds__(128) void gemm_bf16_wmma(
    const bf16_t* __restrict__ A, const bf16_t* __restrict__ W,
    float* __restrict__ Cf, bf16_t* __restrict__ Cb,
    int M, int N, int K) {
  const int wave   = blockIdx.x * (blockDim.x >> 5) + (threadIdx.x >> 5);
  const int ntiles = N >> 6;
  const int mt     = wave / ntiles;
  const int nt     = wave - mt * ntiles;
  if (mt * 64 >= M) return;

  const bf16_t* Ab = A + (size_t)(mt * 64) * (size_t)K;
  const bf16_t* Wb = W + (size_t)(nt * 64) * (size_t)K;

  v8f c[4][4] = {};

  for (int k0 = 0; k0 < K; k0 += 32) {
    v16bf fa[4], fb[4];
#pragma unroll
    for (int i = 0; i < 4; ++i) fa[i] = ld_frag(Ab + (size_t)(i * 16) * (size_t)K, K, k0);
#pragma unroll
    for (int j = 0; j < 4; ++j) fb[j] = ld_frag(Wb + (size_t)(j * 16) * (size_t)K, K, k0);
#pragma unroll
    for (int i = 0; i < 4; ++i)
#pragma unroll
      for (int j = 0; j < 4; ++j)
        c[i][j] = wmma_bf16(fa[i], fb[j], c[i][j]);
  }

  const int lane = threadIdx.x & 31;
  const int nn   = lane & 15;
  const int mb   = (lane >> 4) << 3;
#pragma unroll
  for (int i = 0; i < 4; ++i)
#pragma unroll
    for (int j = 0; j < 4; ++j)
#pragma unroll
      for (int r = 0; r < 8; ++r) {
        const size_t row = (size_t)(mt * 64 + i * 16 + mb + r);
        const size_t col = (size_t)(nt * 64 + j * 16 + nn);
        const float  v   = c[i][j][r];
        if (STORE_BF16) {
          if (TRANSP) Cb[col * (size_t)M + row] = (bf16_t)v;
          else        Cb[row * (size_t)N + col] = (bf16_t)v;
        } else {
          Cf[row * (size_t)N + col] = v;
        }
      }
}

// ---------------------------------------------------------------------------
// 4. Flash attention + sigmoid gate.
//    Block = 4 waves sharing one (b, head); each wave owns a 16-query tile.
//    K (keys x hd) and V^T (hd x keys) tiles double-buffered in LDS via the
//    Tensor Data Mover; TDM for tile i+1 overlaps WMMA on tile i
//    (s_wait_tensorcnt 2 leaves the prefetch in flight).
//    qg: (B*S, 4096) bf16 [q | gate]; kbuf: (B*S, 512) bf16;
//    vt: (NKV*HD, B*S) bf16 (V transposed); ob: (B*S, 2048) bf16
// ---------------------------------------------------------------------------
__global__ __launch_bounds__(128) void attn_flash_wmma(
    const bf16_t* __restrict__ qg, const bf16_t* __restrict__ kbuf,
    const bf16_t* __restrict__ vt, bf16_t* __restrict__ ob) {
  __shared__ __align__(16) bf16_t Kt[2][32][HD];     // 2 x 8 KB key tiles
  __shared__ __align__(16) bf16_t Vt[2][HD][32];     // 2 x 8 KB V^T tiles
  __shared__ __align__(16) float  sc[4][16][32];     // scores (per wave)
  __shared__ __align__(16) bf16_t pb[4][16][32];     // exp(scores) bf16 (per wave)
  __shared__ float alpha_s[4][16];
  __shared__ float lrow[4][16];

  const int w    = threadIdx.x >> 5;
  const int lane = threadIdx.x & 31;
  // gridDim.x = B * NH * (S/64) = 1024
  const int q64  = blockIdx.x & 31;          // 64-query group
  const int h    = (blockIdx.x >> 5) & 15;
  const int b    = blockIdx.x >> 9;
  const int kv   = h >> 2;                   // G = 4
  const int qt   = q64 * 4 + w;              // this wave's 16-row tile

  const bf16_t* Q   = qg  + ((size_t)(b * S_LEN) + (size_t)qt * 16) * QG_N + h * HD;
  const bf16_t* Gt  = Q + NH * HD;           // gate half
  const bf16_t* Kp  = kbuf + (size_t)(b * S_LEN) * KV_N + kv * HD;
  const bf16_t* Vp0 = vt + (size_t)(kv * HD) * BS + (size_t)(b * S_LEN);

  v16bf qa[4];
#pragma unroll
  for (int i = 0; i < 4; ++i) qa[i] = ld_frag(Q, QG_N, 32 * i);

  v8f acc[8] = {};
  float mi = -1e30f, li = 0.f;
  const float scale = 0.08838834764831845f;  // 1/sqrt(128)

  const int nn = lane & 15;
  const int mb = (lane >> 4) << 3;

  const int NT = S_LEN / 32;                 // 64 key tiles
  if (w == 0) {                              // prologue: stage tile 0
    tdm_load_tile(&Kt[0][0][0], Kp,  HD, 32, KV_N);
    tdm_load_tile(&Vt[0][0][0], Vp0, 32, HD, BS);
  }

  for (int it = 0; it < NT; ++it) {
    const int buf = it & 1;
    if (w == 0) {
      if (it + 1 < NT) {                     // prefetch tile it+1 into other buffer
        tdm_load_tile(&Kt[buf ^ 1][0][0], Kp + (size_t)(it + 1) * 32 * KV_N, HD, 32, KV_N);
        tdm_load_tile(&Vt[buf ^ 1][0][0], Vp0 + (it + 1) * 32, 32, HD, BS);
        tdm_wait_le2();                      // tile it complete; prefetch in flight
      } else {
        tdm_wait0();
      }
    }
    __syncthreads();

    // ---- scores: S[16q x 32keys] = Q (16x128) @ K^T (from LDS) ----
    v8f s0 = {}, s1 = {};
#pragma unroll
    for (int i = 0; i < 4; ++i) {
      v16bf fk0 = ld_frag(&Kt[buf][0][0],  HD, 32 * i);
      s0 = wmma_bf16(qa[i], fk0, s0);
      v16bf fk1 = ld_frag(&Kt[buf][16][0], HD, 32 * i);
      s1 = wmma_bf16(qa[i], fk1, s1);
    }
#pragma unroll
    for (int r = 0; r < 8; ++r) {
      sc[w][mb + r][nn]      = s0[r] * scale;
      sc[w][mb + r][nn + 16] = s1[r] * scale;
    }
    __builtin_amdgcn_wave_barrier();

    // ---- online softmax (lanes 0..15 each own one query row) ----
    if (lane < 16) {
      const int m = lane;
      float tmax = -1e30f;
#pragma unroll
      for (int j = 0; j < 32; ++j) tmax = fmaxf(tmax, sc[w][m][j]);
      const float nm = fmaxf(mi, tmax);
      const float al = __expf(mi - nm);
      float sum = 0.f;
#pragma unroll
      for (int j = 0; j < 32; ++j) {
        const float pz = __expf(sc[w][m][j] - nm);
        sum += pz;
        pb[w][m][j] = (bf16_t)pz;
      }
      li = al * li + sum;
      mi = nm;
      alpha_s[w][m] = al;
    }
    __builtin_amdgcn_wave_barrier();

    // ---- rescale running output by alpha ----
#pragma unroll
    for (int t = 0; t < 8; ++t)
#pragma unroll
      for (int r = 0; r < 8; ++r) acc[t][r] *= alpha_s[w][mb + r];

    // ---- O += P(16x32) @ V(32x128); V^T rows contiguous -> b128 ds loads ----
    const v16bf pa = ld_frag(&pb[w][0][0], 32, 0);
#pragma unroll
    for (int t = 0; t < 8; ++t) {
      v16bf fv = ld_frag(&Vt[buf][t * 16][0], 32, 0);
      acc[t] = wmma_bf16(pa, fv, acc[t]);
    }
    __syncthreads();   // all waves done with buf before it is overwritten
  }

  if (lane < 16) lrow[w][lane] = 1.0f / li;
  __builtin_amdgcn_wave_barrier();

  // ---- epilogue: normalize, sigmoid-gate, store bf16 (b,s, h*HD+d) ----
  const size_t orow0 = (size_t)(b * S_LEN) + (size_t)qt * 16;
#pragma unroll
  for (int t = 0; t < 8; ++t)
#pragma unroll
    for (int r = 0; r < 8; ++r) {
      const int   m  = mb + r;
      const float ov = acc[t][r] * lrow[w][m];
      const float gv = (float)Gt[(size_t)m * QG_N + t * 16 + nn];
      const float gs = 1.0f / (1.0f + __expf(-gv));
      ob[(orow0 + m) * (NH * HD) + h * HD + t * 16 + nn] = (bf16_t)(ov * gs);
    }
}

// ---------------------------------------------------------------------------
// Launch
// ---------------------------------------------------------------------------
extern "C" void kernel_launch(void* const* d_in, const int* in_sizes, int n_in,
                              void* d_out, int out_size, void* d_ws, size_t ws_size,
                              hipStream_t stream) {
  const float* hs = (const float*)d_in[0];
  // d_in[1] = position_ids (unused by the reference computation)
  const float* Wq = (const float*)d_in[2];
  const float* Wk = (const float*)d_in[3];
  const float* Wv = (const float*)d_in[4];
  const float* Wo = (const float*)d_in[5];
  const float* nw = (const float*)d_in[6];
  float* out = (float*)d_out;

  const size_t M = (size_t)BS;  // 4096 rows

  // workspace layout (bf16 buffers), 256B aligned offsets
  char* ws = (char*)d_ws;
  size_t off = 0;
  auto take = [&](size_t bytes) { char* p = ws + off; off = (off + bytes + 255) & ~(size_t)255; return p; };
  bf16_t* x_bf  = (bf16_t*)take(M * HID * 2);
  bf16_t* wq_bf = (bf16_t*)take((size_t)QG_N * HID * 2);
  bf16_t* wk_bf = (bf16_t*)take((size_t)KV_N * HID * 2);
  bf16_t* wv_bf = (bf16_t*)take((size_t)KV_N * HID * 2);
  bf16_t* wo_bf = (bf16_t*)take((size_t)HID * HID * 2);
  bf16_t* qg_bf = (bf16_t*)take(M * QG_N * 2);
  bf16_t* k_bf  = (bf16_t*)take(M * KV_N * 2);
  bf16_t* vt_bf = (bf16_t*)take((size_t)KV_N * M * 2);   // V^T: (NKV*HD) x (B*S)
  bf16_t* o_bf  = (bf16_t*)take(M * HID * 2);
  (void)ws_size; (void)in_sizes; (void)n_in; (void)out_size;

  // 1. RMSNorm -> bf16
  rmsnorm_bf16_kernel<<<(int)M, 256, 0, stream>>>(hs, nw, x_bf);

  // 2. weight conversions
  f32_to_bf16_kernel<<<2048, 256, 0, stream>>>(Wq, wq_bf, (long long)QG_N * HID);
  f32_to_bf16_kernel<<<512,  256, 0, stream>>>(Wk, wk_bf, (long long)KV_N * HID);
  f32_to_bf16_kernel<<<512,  256, 0, stream>>>(Wv, wv_bf, (long long)KV_N * HID);
  f32_to_bf16_kernel<<<1024, 256, 0, stream>>>(Wo, wo_bf, (long long)HID * HID);

  // 3. projections: qg = x@Wq^T, k = x@Wk^T, v^T = (x@Wv^T)^T
  gemm_bf16_wmma<1, 0><<<(64 * 64) / 4, 128, 0, stream>>>(x_bf, wq_bf, nullptr, qg_bf,
                                                          (int)M, QG_N, HID);
  gemm_bf16_wmma<1, 0><<<(64 * 8) / 4, 128, 0, stream>>>(x_bf, wk_bf, nullptr, k_bf,
                                                         (int)M, KV_N, HID);
  gemm_bf16_wmma<1, 1><<<(64 * 8) / 4, 128, 0, stream>>>(x_bf, wv_bf, nullptr, vt_bf,
                                                         (int)M, KV_N, HID);

  // 4. flash attention + gate: B * NH * (S/64) = 1024 blocks of 4 waves
  attn_flash_wmma<<<1024, 128, 0, stream>>>(qg_bf, k_bf, vt_bf, o_bf);

  // 5. output projection (fp32 out): out = o @ Wo^T
  gemm_bf16_wmma<0, 0><<<(64 * 32) / 4, 128, 0, stream>>>(o_bf, wo_bf, out, nullptr,
                                                          (int)M, HID, HID);
}
